// Decoder_14302241095776
// MI455X (gfx1250) — matrix operand order repaired and checked by reading, not compiled, not run on hardware
//
#include <hip/hip_runtime.h>

#define BB 256
#define TT 4096
#define CC 8
#define HH 128

typedef _Float16 half_t;
typedef _Float16 v16h __attribute__((ext_vector_type(16)));
typedef _Float16 v8h  __attribute__((ext_vector_type(8)));
typedef float    v8f  __attribute__((ext_vector_type(8)));

// Split workgroup barrier: only LDS (dscnt) is fenced before signaling, so
// global stores / loads can overlap barrier latency between signal and wait.
#define BAR_SIGNAL() asm volatile("s_wait_dscnt 0x0\n\ts_barrier_signal -1" ::: "memory")
#define BAR_WAIT()   asm volatile("s_barrier_wait -1" ::: "memory")

__device__ __forceinline__ float fast_tanh(float x) {
    // tanh(x) = (e-1)/(e+1), e = exp2(2x*log2e); clamp avoids inf/inf -> NaN
    x = fminf(fmaxf(x, -10.0f), 10.0f);
    float e = __builtin_amdgcn_exp2f(x * 2.8853900817779268f);
    return (e - 1.0f) * __builtin_amdgcn_rcpf(e + 1.0f);
}

__device__ __forceinline__ v8f wmma_f16(v16h a, v16h b, v8f c) {
    // D = A(16x32 f16) * B(32x16 f16) + C(16x16 f32)
    return __builtin_amdgcn_wmma_f32_16x16x32_f16(
        /*neg_a=*/false, a, /*neg_b=*/false, b,
        /*c_mod=*/(short)0, c, /*reuse_a=*/false, /*reuse_b=*/false);
}

// Load a 16x32 f16 A-fragment chunk of h from LDS (row-major 16 x HH halves).
// A layout (contiguous K, matching B): lane<16 holds row=lane, K={0..15} in halves0-15;
// lane>=16 holds row=lane-16, K={16..31}.
__device__ __forceinline__ v16h load_A_chunk(const half_t* hp, int row, int kc, int hi) {
    const int c0 = kc * 32 + hi * 16;
    v8h lo = *(const v8h*)(hp + row * HH + c0);
    v8h hh = *(const v8h*)(hp + row * HH + c0 + 8);
    v16h a;
#pragma unroll
    for (int j = 0; j < 8; ++j) { a[j] = lo[j]; a[j + 8] = hh[j]; }
    return a;
}

// Two overlapped accumulation chains over the 4 K-chunks, combined at the end.
__device__ __forceinline__ v8f hidden_matmul(const half_t* hp, int row, int hi,
                                             const v16h* Bw, v8f accA) {
    v16h A0 = load_A_chunk(hp, row, 0, hi);
    v16h A1 = load_A_chunk(hp, row, 1, hi);
    v16h A2 = load_A_chunk(hp, row, 2, hi);
    v16h A3 = load_A_chunk(hp, row, 3, hi);
    v8f accB;
#pragma unroll
    for (int r = 0; r < 8; ++r) accB[r] = 0.0f;
    accA = wmma_f16(A0, Bw[0], accA);
    accB = wmma_f16(A2, Bw[2], accB);
    accA = wmma_f16(A1, Bw[1], accA);
    accB = wmma_f16(A3, Bw[3], accB);
#pragma unroll
    for (int r = 0; r < 8; ++r) accA[r] += accB[r];
    return accA;
}

// ---------------- Pass 1: u = tanh(x*W1 + b1), stored f16 [B,T,C] ----------------
__global__ void __launch_bounds__(256) u_kernel(const float* __restrict__ x,
                                                const float* __restrict__ W1,
                                                const float* __restrict__ b1,
                                                half_t* __restrict__ u) {
    int i = blockIdx.x * 256 + threadIdx.x;   // one (b,t) per thread, B*T total
    float xv = x[i];
    v8h r;
#pragma unroll
    for (int c = 0; c < CC; ++c)
        r[c] = (half_t)fast_tanh(xv * W1[c] + b1[c]);
    *(v8h*)(u + (size_t)i * CC) = r;
}

// ---------------- Pass 2: sequential RNN scan + output projection ----------------
// grid = B/16 workgroups; block = 288 threads (9 wave32 waves).
// waves 0..7: recurrence N-tiles (16 cols each). wave 8: output projection (lags 1 step).
__global__ void __launch_bounds__(288) scan_kernel(const half_t* __restrict__ u,
                                                   const float* __restrict__ W_ih,
                                                   const float* __restrict__ b_ih,
                                                   const float* __restrict__ W_hh,
                                                   const float* __restrict__ b_hh,
                                                   const float* __restrict__ W2,
                                                   const float* __restrict__ b2,
                                                   float* __restrict__ out) {
    __shared__ __align__(16) half_t hb[2][16 * HH];   // double-buffered h tile, f16

    const int w    = threadIdx.x >> 5;
    const int lane = threadIdx.x & 31;
    const int hi   = lane >> 4;     // half-wave select
    const int n    = lane & 15;     // column within tile / row for A loads
    const int b0   = blockIdx.x * 16;

    // h_{-1} = 0
    for (int i = threadIdx.x; i < 16 * HH; i += 288) hb[0][i] = (half_t)0.0f;

    if (w < 8) {
        // ---------------- recurrence waves ----------------
        const int ncol = w * 16 + n;   // output column of h this lane produces

        // Resident B-fragments of W_hh^T: B(K=k,N=n) = W_hh[n][k]. 4 K-chunks of 32.
        v16h Bh[4];
#pragma unroll
        for (int kc = 0; kc < 4; ++kc)
#pragma unroll
            for (int j = 0; j < 16; ++j)
                Bh[kc][j] = (half_t)W_hh[ncol * HH + kc * 32 + hi * 16 + j];

        // B-fragment of W_ih (K=c, zero-padded past C=8): B(K=c,N=n) = W_ih[n][c]
        v16h Bi;
#pragma unroll
        for (int j = 0; j < 16; ++j) {
            int c = hi * 16 + j;
            Bi[j] = (c < CC) ? (half_t)W_ih[ncol * CC + c] : (half_t)0.0f;
        }

        const float bias = b_ih[ncol] + b_hh[ncol];
        const half_t* urow = u + ((size_t)(b0 + n) * TT) * CC;  // lanes<16 only

        // A-fragment of u tile [16 x 8], zero-padded to K=32; halves 8..15
        // stay zero forever, halves 0..7 are refreshed each step (lanes<16).
        v16h Au;
#pragma unroll
        for (int j = 0; j < 16; ++j) Au[j] = (half_t)0.0f;
        if (!hi) {
            v8h uv = *(const v8h*)(urow);
#pragma unroll
            for (int j = 0; j < 8; ++j) Au[j] = uv[j];
        }

        BAR_SIGNAL();
        BAR_WAIT();
        int p = 0;
        for (int t = 0; t < TT; ++t) {
            v8f acc;
#pragma unroll
            for (int r = 0; r < 8; ++r) acc[r] = bias;
            acc = wmma_f16(Au, Bi, acc);                   // z_t (independent of LDS)
            acc = hidden_matmul(hb[p], n, hi, Bh, acc);    // + h_{t-1} @ W_hh^T

            // h_t = tanh(acc), pack f16 into the other buffer
            half_t* hq = hb[p ^ 1];
#pragma unroll
            for (int r = 0; r < 8; ++r)
                hq[(r + hi * 8) * HH + ncol] = (half_t)fast_tanh(acc[r]);

            BAR_SIGNAL();
            // overlap barrier latency with next step's u load (global)
            if (!hi && (t + 1) < TT) {
                v8h uv = *(const v8h*)(urow + (size_t)(t + 1) * CC);
#pragma unroll
                for (int j = 0; j < 8; ++j) Au[j] = uv[j];
                if (t + 17 < TT)
                    __builtin_prefetch(urow + (size_t)(t + 17) * CC, 0, 0);
            }
            BAR_WAIT();
            p ^= 1;
        }
    } else {
        // ---------------- output-projection wave (lags one step) ----------------
        // B(K=k,N=c) = W2[c][k], columns >= 8 are don't-care
        v16h Bo[4];
#pragma unroll
        for (int kc = 0; kc < 4; ++kc)
#pragma unroll
            for (int j = 0; j < 16; ++j)
                Bo[kc][j] = (n < CC) ? (half_t)W2[n * HH + kc * 32 + hi * 16 + j]
                                     : (half_t)0.0f;
        const float bias = (n < CC) ? b2[n] : 0.0f;

        BAR_SIGNAL();
        BAR_WAIT();
        int p = 0;
        for (int t = 0; t < TT; ++t) {
            v8f acc;
            if (t > 0) {
#pragma unroll
                for (int r = 0; r < 8; ++r) acc[r] = bias;
                acc = hidden_matmul(hb[p], n, hi, Bo, acc);   // reads h_{t-1}
            }
            BAR_SIGNAL();
            // tanh + global stores overlap the barrier; nobody reads `out`
            if (t > 0 && n < CC) {
#pragma unroll
                for (int r = 0; r < 8; ++r)
                    out[((size_t)(b0 + r + hi * 8) * TT + (t - 1)) * CC + n] =
                        fast_tanh(acc[r]);
            }
            BAR_WAIT();
            p ^= 1;
        }
        // final step: output for t = T-1 from hb[p]
        {
            v8f acc;
#pragma unroll
            for (int r = 0; r < 8; ++r) acc[r] = bias;
            acc = hidden_matmul(hb[p], n, hi, Bo, acc);
#pragma unroll
            for (int r = 0; r < 8; ++r)
                if (n < CC)
                    out[((size_t)(b0 + r + hi * 8) * TT + (TT - 1)) * CC + n] =
                        fast_tanh(acc[r]);
        }
    }
}

extern "C" void kernel_launch(void* const* d_in, const int* in_sizes, int n_in,
                              void* d_out, int out_size, void* d_ws, size_t ws_size,
                              hipStream_t stream) {
    const float* x    = (const float*)d_in[0];
    const float* W1   = (const float*)d_in[1];
    const float* b1   = (const float*)d_in[2];
    const float* W_ih = (const float*)d_in[3];
    const float* b_ih = (const float*)d_in[4];
    const float* W_hh = (const float*)d_in[5];
    const float* b_hh = (const float*)d_in[6];
    const float* W2   = (const float*)d_in[7];
    const float* b2   = (const float*)d_in[8];

    half_t* u = (half_t*)d_ws;   // B*T*C f16 = 16 MB staged activations

    u_kernel<<<(BB * TT) / 256, 256, 0, stream>>>(x, W1, b1, u);
    scan_kernel<<<BB / 16, 288, 0, stream>>>(u, W_ih, b_ih, W_hh, b_hh, W2, b2,
                                             (float*)d_out);
}
